// CapsuleNetwork_73727408603530
// MI455X (gfx1250) — matrix operand order repaired and checked by reading, not compiled, not run on hardware
//
#include <hip/hip_runtime.h>
#include <hip/hip_bf16.h>

typedef __attribute__((ext_vector_type(16))) __bf16          v16bf;
typedef __attribute__((ext_vector_type(16))) unsigned short  v16u;
typedef __attribute__((ext_vector_type(8)))  float           v8f;

#define S_LEN 200
#define D_DIM 64
#define NI    4
#define K_DIM 256      // NI * D
#define BT    16       // batch tile per workgroup (matches WMMA M)
#define A_STR 72       // padded LDS row strides (avoid bank conflicts, keep 16B align: 72*2=144B)
#define B_STR 72
#define H_STR 264      // hat row stride in floats

__device__ __forceinline__ unsigned short f2bf(float x) {
  union { float f; unsigned int u; } v; v.f = x;
  unsigned int u = v.u;
  u += 0x7fffu + ((u >> 16) & 1u);   // round-to-nearest-even
  return (unsigned short)(u >> 16);
}

extern "C" __global__ __launch_bounds__(256)
void capsule_routing_kernel(const float* __restrict__ his,   // [B,S,D]
                            const float* __restrict__ eb,    // [B,D]
                            const float* __restrict__ mask,  // [B,S]
                            const float* __restrict__ w,     // [S,K,D]
                            float* __restrict__ out,         // [B*NI*D] ++ [B*D]
                            int B)
{
  extern __shared__ char smem[];
  unsigned short* Abf = (unsigned short*)smem;          // [BT][A_STR] bf16
  unsigned short* Bbf = Abf + BT * A_STR;               // [K_DIM][B_STR] bf16
  float* hat = (float*)(Bbf + K_DIM * B_STR);           // [BT][H_STR]
  float* cw  = hat + BT * H_STR;                        // [BT][NI][S_LEN]
  float* mk  = cw  + BT * NI * S_LEN;                   // [BT][S_LEN]
  float* ic  = mk  + BT * S_LEN;                        // [BT][NI][D_DIM]
  float* cap = ic  + BT * NI * D_DIM;                   // [BT][NI][D_DIM]
  float* swb = cap + BT * NI * D_DIM;                   // [BT][NI]
  float* scl = swb + BT * NI;                           // [BT][NI]

  const int tid  = threadIdx.x;
  const int lane = tid & 31;
  const int wv   = tid >> 5;                            // 8 wave32s
  const int b0   = blockIdx.x * BT;

  // init capsule weights to zero, stage mask tile
  for (int i = tid; i < BT * NI * S_LEN; i += 256) cw[i] = 0.f;
  for (int i = tid; i < BT * S_LEN; i += 256) {
    int b = i / S_LEN, s = i % S_LEN;
    mk[i] = mask[(size_t)(b0 + b) * S_LEN + s];
  }
  __syncthreads();

  for (int pass = 0; pass < 3; ++pass) {
    for (int i = tid; i < BT * NI * D_DIM; i += 256) ic[i] = 0.f;
    __syncthreads();

    for (int s = 0; s < S_LEN; ++s) {
      // ---- stage A = his[b0..b0+15, s, :] as bf16 (16x64) ----
      {
        const int r  = tid >> 4;
        const int c4 = (tid & 15) << 2;
        const float4 a4 = *(const float4*)(his + ((size_t)(b0 + r) * S_LEN + s) * D_DIM + c4);
        unsigned short* dst = Abf + r * A_STR + c4;
        dst[0] = f2bf(a4.x); dst[1] = f2bf(a4.y); dst[2] = f2bf(a4.z); dst[3] = f2bf(a4.w);
      }
      // ---- stage B = w[s] as bf16 (256x64); one k-row per thread ----
      {
        const float* wr = w + ((size_t)s * K_DIM + tid) * D_DIM;
        unsigned short* dst = Bbf + tid * B_STR;
#pragma unroll
        for (int c = 0; c < D_DIM; c += 4) {
          const float4 b4 = *(const float4*)(wr + c);
          dst[c+0] = f2bf(b4.x); dst[c+1] = f2bf(b4.y);
          dst[c+2] = f2bf(b4.z); dst[c+3] = f2bf(b4.w);
        }
        if (s + 1 < S_LEN) __builtin_prefetch(wr + (size_t)K_DIM * D_DIM, 0, 1);
      }
      __syncthreads();

      // ---- WMMA: hat_tile(16x256) = A(16x64) @ W[s]^T ; wave wv owns k-tiles 2wv,2wv+1 ----
#pragma unroll
      for (int t = 0; t < 2; ++t) {
        const int k0 = (wv * 2 + t) * 16;
        v8f c = {0.f, 0.f, 0.f, 0.f, 0.f, 0.f, 0.f, 0.f};
#pragma unroll
        for (int c32 = 0; c32 < 2; ++c32) {
          const int hi = (lane >> 4) & 1;
          union { uint4 q[2]; v16u v; } af, bf;
          // A frag: lane row = lane%16; 16 contiguous K at ak = c32*32 + hi*16
          const unsigned short* ap = Abf + (lane & 15) * A_STR + (c32 * 32 + hi * 16);
          af.q[0] = *(const uint4*)(ap);
          af.q[1] = *(const uint4*)(ap + 8);
          // B frag: lane col = k0 + lane%16; 16 contiguous K at bk = c32*32 + hi*16
          const unsigned short* bp = Bbf + (size_t)(k0 + (lane & 15)) * B_STR + (c32 * 32 + hi * 16);
          bf.q[0] = *(const uint4*)(bp);
          bf.q[1] = *(const uint4*)(bp + 8);
          c = __builtin_amdgcn_wmma_f32_16x16x32_bf16(
                false, __builtin_bit_cast(v16bf, af.v),
                false, __builtin_bit_cast(v16bf, bf.v),
                (short)0, c, false, false);
        }
        const int m0  = ((lane >> 4) & 1) ? 8 : 0;
        const int col = k0 + (lane & 15);
#pragma unroll
        for (int r = 0; r < 8; ++r) hat[(r + m0) * H_STR + col] = c[r];
      }
      __syncthreads();

      // ---- routing R1: delta = <hat, cap>, cw += delta (passes 1,2) ----
      if (pass > 0 && tid < BT * NI) {
        const int b = tid >> 2, n = tid & 3;
        const float* hr = hat + b * H_STR + n * D_DIM;
        const float* cr = cap + (b * NI + n) * D_DIM;
        float d = 0.f;
#pragma unroll 8
        for (int k = 0; k < D_DIM; ++k) d += hr[k] * cr[k];
        cw[(b * NI + n) * S_LEN + s] += d;
      }
      __syncthreads();
      // ---- softmax over NI interests, masked ----
      if (tid < BT * NI) {
        const int b = tid >> 2, n = tid & 3;
        float sw;
        if (pass == 0) {
          sw = 0.25f;                                  // softmax of zeros
        } else {
          const float* c0 = cw + b * NI * S_LEN + s;
          const float x0 = c0[0], x1 = c0[S_LEN], x2 = c0[2 * S_LEN], x3 = c0[3 * S_LEN];
          const float m  = fmaxf(fmaxf(x0, x1), fmaxf(x2, x3));
          const float e0 = __expf(x0 - m), e1 = __expf(x1 - m),
                      e2 = __expf(x2 - m), e3 = __expf(x3 - m);
          const float en = (n == 0) ? e0 : (n == 1) ? e1 : (n == 2) ? e2 : e3;
          sw = en / (e0 + e1 + e2 + e3);
        }
        if (mk[b * S_LEN + s] == 0.f) sw = 0.f;
        swb[b * NI + n] = sw;
      }
      __syncthreads();
      // ---- R2: ic += sw * hat ----
      {
        const int b = tid >> 4, dq = (tid & 15) << 2;
#pragma unroll
        for (int n = 0; n < NI; ++n) {
          const float swv = swb[b * NI + n];
          float* icp = ic + (b * NI + n) * D_DIM + dq;
          const float* hp = hat + b * H_STR + n * D_DIM + dq;
          icp[0] += swv * hp[0]; icp[1] += swv * hp[1];
          icp[2] += swv * hp[2]; icp[3] += swv * hp[3];
        }
      }
      __syncthreads();
    } // s

    // ---- squash: cap = |ic|^2/(1+|ic|^2)/sqrt(|ic|^2+1e-9) * ic ----
    if (tid < BT * NI) {
      const int b = tid >> 2, n = tid & 3;
      const float* icp = ic + (b * NI + n) * D_DIM;
      float norm = 0.f;
#pragma unroll 8
      for (int k = 0; k < D_DIM; ++k) norm += icp[k] * icp[k];
      scl[b * NI + n] = norm / (1.f + norm) / sqrtf(norm + 1e-9f);
    }
    __syncthreads();
    {
      const int b = tid >> 4, dq = (tid & 15) << 2;
#pragma unroll
      for (int n = 0; n < NI; ++n) {
        const float sc = scl[b * NI + n];
        float* cp = cap + (b * NI + n) * D_DIM + dq;
        const float* icp = ic + (b * NI + n) * D_DIM + dq;
        cp[0] = sc * icp[0]; cp[1] = sc * icp[1];
        cp[2] = sc * icp[2]; cp[3] = sc * icp[3];
      }
    }
    __syncthreads();
  } // pass

  // ---- outputs: interest_capsule then readout (argmax(softmax)==argmax, first-max tie-break) ----
  {
    const int b = tid >> 4, dq = (tid & 15) << 2;
#pragma unroll
    for (int n = 0; n < NI; ++n) {
      const float* cp = cap + (b * NI + n) * D_DIM + dq;
      float* op = out + ((size_t)(b0 + b) * NI + n) * D_DIM + dq;
      op[0] = cp[0]; op[1] = cp[1]; op[2] = cp[2]; op[3] = cp[3];
    }
  }
  if (tid < BT) {
    const int b = tid;
    const float* e = eb + (size_t)(b0 + b) * D_DIM;
    float best = -3.0e38f; int bi = 0;
#pragma unroll
    for (int n = 0; n < NI; ++n) {
      const float* cp = cap + (b * NI + n) * D_DIM;
      float a = 0.f;
      for (int k = 0; k < D_DIM; ++k) a += cp[k] * e[k];
      if (a > best) { best = a; bi = n; }
    }
    float* op = out + (size_t)B * NI * D_DIM + (size_t)(b0 + b) * D_DIM;
    const float* cp = cap + (b * NI + bi) * D_DIM;
    for (int k = 0; k < D_DIM; ++k) op[k] = cp[k];
  }
}

extern "C" void kernel_launch(void* const* d_in, const int* in_sizes, int n_in,
                              void* d_out, int out_size, void* d_ws, size_t ws_size,
                              hipStream_t stream) {
  (void)n_in; (void)out_size; (void)d_ws; (void)ws_size;
  const float* his  = (const float*)d_in[0];
  const float* eb   = (const float*)d_in[1];
  const float* mask = (const float*)d_in[2];
  const float* w    = (const float*)d_in[3];
  float* out = (float*)d_out;

  const int B = in_sizes[0] / (S_LEN * D_DIM);   // 2048
  const size_t smem =
      (size_t)(BT * A_STR + K_DIM * B_STR) * 2 +
      (size_t)(BT * H_STR + BT * NI * S_LEN + BT * S_LEN +
               2 * BT * NI * D_DIM + 2 * BT * NI) * 4;   // ~150 KB < 320 KB/WGP

  capsule_routing_kernel<<<dim3(B / BT), dim3(256), smem, stream>>>(
      his, eb, mask, w, out, B);
}
